// MultiHeadedAttention_24541443129837
// MI455X (gfx1250) — matrix operand order repaired and checked by reading, not compiled, not run on hardware
//
#include <hip/hip_runtime.h>

// ---------------------------------------------------------------------------
// MI455X (gfx1250) fused multi-head attention: bf16 WMMA + TDM async staging.
// ---------------------------------------------------------------------------

typedef __attribute__((ext_vector_type(16))) __bf16 v16bf;
typedef __attribute__((ext_vector_type(8)))  __bf16 v8bf;
typedef __attribute__((ext_vector_type(8)))  float  v8f;
typedef __attribute__((ext_vector_type(4)))  unsigned int v4u;
typedef __attribute__((ext_vector_type(8)))  int v8i;
typedef __attribute__((ext_vector_type(4)))  int v4i;

#define BATCH  2
#define SEQ    2048
#define DMODEL 1024
#define NHEADS 16
#define DKH    64

#if defined(__gfx1250__) && __has_builtin(__builtin_amdgcn_tensor_load_to_lds) && \
    __has_builtin(__builtin_amdgcn_s_wait_tensorcnt)
#define USE_TDM 1
#else
#define USE_TDM 0
#endif

static __device__ inline v8f wmma_bf16f32(v16bf a, v16bf b, v8f c) {
  return __builtin_amdgcn_wmma_f32_16x16x32_bf16(
      /*neg_a=*/false, a, /*neg_b=*/false, b,
      /*c_mod=*/(short)0, c, /*reuse_a=*/false, /*reuse_b=*/false);
}

// A fragment (16x32 bf16) from row-major bf16 memory (global or LDS).
static __device__ inline v16bf load_a_bf16(const __bf16* __restrict__ src, int ld) {
  const int lane = threadIdx.x & 31;
  const int row  = lane & 15;
  const int hl   = (lane >> 4) & 1;
  const __bf16* p = src + (size_t)row * ld + 8 * hl;
  const v8bf lo = *reinterpret_cast<const v8bf*>(p);
  const v8bf hi = *reinterpret_cast<const v8bf*>(p + 16);
  return __builtin_shufflevector(lo, hi, 0,1,2,3,4,5,6,7,8,9,10,11,12,13,14,15);
}

// B fragment (32x16), B[k][n] = src[n*ld + k], bf16 memory (contiguous in k).
static __device__ inline v16bf load_bt_bf16(const __bf16* __restrict__ src, int ld) {
  const int lane = threadIdx.x & 31;
  const int n  = lane & 15;
  const int kb = (lane >> 4) << 3;
  const __bf16* p = src + (size_t)n * ld + kb;
  const v8bf lo = *reinterpret_cast<const v8bf*>(p);
  const v8bf hi = *reinterpret_cast<const v8bf*>(p + 16);
  return __builtin_shufflevector(lo, hi, 0,1,2,3,4,5,6,7,8,9,10,11,12,13,14,15);
}

#if USE_TDM
// Build a D# and issue TENSOR_LOAD_TO_LDS for a 2D bf16 tile.
// gptr points at the tile start; tile0 contiguous elems, tile1 rows (0/1 => 1D),
// stride0 = row stride in elements.  (CDNA5 ISA)
static __device__ inline void tdm_load_2d(unsigned lds_byte, const void* gptr,
                                          unsigned tile0, unsigned tile1,
                                          unsigned td0, unsigned td1,
                                          unsigned long long stride0) {
  const unsigned long long ga = (unsigned long long)gptr;
  v4u g0;
  g0[0] = 1u;                                            // count=1, user mode
  g0[1] = lds_byte;                                      // lds_addr
  g0[2] = (unsigned)(ga & 0xFFFFFFFFu);                  // global_addr[31:0]
  g0[3] = (unsigned)((ga >> 32) & 0x1FFFFFFu) | (2u << 30);  // addr[56:32], type=2
  v8i g1;
  g1[0] = (int)(1u << 16);                               // data_size = 1 (2 bytes)
  g1[1] = (int)((td0 & 0xFFFFu) << 16);                  // tensor_dim0[15:0]
  g1[2] = (int)(((td0 >> 16) & 0xFFFFu) | ((td1 & 0xFFFFu) << 16));
  g1[3] = (int)(((td1 >> 16) & 0xFFFFu) | ((tile0 & 0xFFFFu) << 16));  // tile_dim0
  g1[4] = (int)(tile1 & 0xFFFFu);                        // tile_dim1, tile_dim2=0
  g1[5] = (int)(stride0 & 0xFFFFFFFFu);                  // dim0_stride[31:0]
  g1[6] = (int)((stride0 >> 32) & 0xFFFFu);              // dim0_stride[47:32]
  g1[7] = 0;
  const v4i z4 = {0, 0, 0, 0};
#if __clang_major__ >= 23
  const v8i z8 = {0, 0, 0, 0, 0, 0, 0, 0};
  __builtin_amdgcn_tensor_load_to_lds(g0, g1, z4, z4, z8, 0);
#else
  __builtin_amdgcn_tensor_load_to_lds(g0, g1, z4, z4, 0);
#endif
}

static __device__ inline unsigned lds_byte_addr(const void* p) {
  // Generic pointers into LDS carry the LDS byte offset in their low 32 bits.
  return (unsigned)(unsigned long long)p;
}
#endif  // USE_TDM

// ---------------------------------------------------------------------------
// Kernel 0: bulk f32 -> bf16 conversion (activations + weights, done once).
// ---------------------------------------------------------------------------
__global__ void __launch_bounds__(256)
cvt_f32_to_bf16_kernel(const float* __restrict__ src, __bf16* __restrict__ dst,
                       int n) {
  const int i = (int)(blockIdx.x * blockDim.x + threadIdx.x) * 8;
  if (i + 8 <= n) {
    const float4 a = *reinterpret_cast<const float4*>(src + i);
    const float4 b = *reinterpret_cast<const float4*>(src + i + 4);
    v8bf o;
    o[0]=(__bf16)a.x; o[1]=(__bf16)a.y; o[2]=(__bf16)a.z; o[3]=(__bf16)a.w;
    o[4]=(__bf16)b.x; o[5]=(__bf16)b.y; o[6]=(__bf16)b.z; o[7]=(__bf16)b.w;
    *reinterpret_cast<v8bf*>(dst + i) = o;
  }
}

// ---------------------------------------------------------------------------
// Kernel 1: projection  out = X @ W^T + b  (bf16 in, bf16 head-major out).
// kt_layout==0: out[((b*H+h)*S + s)*Dk + d]   (Q, V)
// kt_layout==1: out[((b*H+h)*Dk + d)*S + s]   (K transposed)
// grid = (M/256, N/64), block = 256 (8 waves), wave -> 32x64 tile
// (2 A fragments x 4 B fragments = 8 WMMAs per k-step; each B feeds 2 WMMAs).
// ---------------------------------------------------------------------------
__global__ void __launch_bounds__(256)
qkv_proj_kernel(const __bf16* __restrict__ X, const __bf16* __restrict__ W,
                const float* __restrict__ bias, __bf16* __restrict__ out,
                const int kt_layout) {
  const int lane = threadIdx.x & 31;
  const int wave = threadIdx.x >> 5;
  const int hl   = (lane >> 4) & 1;
  const int coln = lane & 15;
  const int m0 = blockIdx.x * 256 + wave * 32;
  const int n0 = blockIdx.y * 64;

  v8f acc[8] = {};
  for (int k = 0; k < DMODEL; k += 32) {
    const v16bf a0 = load_a_bf16(X + (size_t)m0 * DMODEL + k, DMODEL);
    const v16bf a1 = load_a_bf16(X + (size_t)(m0 + 16) * DMODEL + k, DMODEL);
    #pragma unroll
    for (int t = 0; t < 4; ++t) {
      const v16bf bfrag =
          load_bt_bf16(W + (size_t)(n0 + 16 * t) * DMODEL + k, DMODEL);
      acc[t]     = wmma_bf16f32(a0, bfrag, acc[t]);
      acc[4 + t] = wmma_bf16f32(a1, bfrag, acc[4 + t]);
    }
  }

  #pragma unroll
  for (int g = 0; g < 2; ++g) {
    #pragma unroll
    for (int t = 0; t < 4; ++t) {
      const int n  = n0 + t * 16 + coln;
      const float bv = bias[n];
      const int hh = n >> 6;
      const int d  = n & 63;
      #pragma unroll
      for (int r = 0; r < 8; ++r) {
        const int m  = m0 + g * 16 + r + 8 * hl;
        const int bb = m >> 11;
        const int s  = m & 2047;
        const float val = acc[g * 4 + t][r] + bv;
        const size_t idx = kt_layout
            ? (((size_t)(bb * NHEADS + hh) * DKH + d) * SEQ + s)
            : (((size_t)(bb * NHEADS + hh) * SEQ + s) * DKH + d);
        out[idx] = (__bf16)val;
      }
    }
  }
}

// ---------------------------------------------------------------------------
// Kernel 2: flash attention per (b,h).  Reference swaps K/V:
//   scores = Q @ Vh^T * 0.125 ; masked softmax ; ctx = P @ Kh
// grid = (S/64, B*H), block = 128 (4 waves), wave -> 16 query rows.
// K/V 32-key blocks staged in LDS by the Tensor Data Mover (double buffered).
// ---------------------------------------------------------------------------
__global__ void __launch_bounds__(128)
attn_kernel(const __bf16* __restrict__ Qh, const __bf16* __restrict__ Kt,
            const __bf16* __restrict__ Vh, const int* __restrict__ mask,
            __bf16* __restrict__ ctx) {
#if USE_TDM
  __shared__ __align__(16) __bf16 vbuf[2][32 * 64];   // [key][d]
  __shared__ __align__(16) __bf16 kbuf[2][64 * 32];   // [d][key]
#endif
  __shared__ __align__(32) __bf16 pstage[4][16 * 32];

  const int lane = threadIdx.x & 31;
  const int wave = threadIdx.x >> 5;
  const int hl   = (lane >> 4) & 1;
  const int coln = lane & 15;
  const int bh = blockIdx.y;
  const int b  = bh >> 4;
  const int h  = bh & 15;
  const int q0 = blockIdx.x * 64 + wave * 16;

  const __bf16* Qbase = Qh + ((size_t)bh * SEQ + q0) * DKH;
  const __bf16* Vbase = Vh + (size_t)bh * SEQ * DKH;     // [s][d]
  const __bf16* Kbase = Kt + (size_t)bh * DKH * SEQ;     // [d][s]
  const int*    mbase = mask + (size_t)b * SEQ * SEQ + (size_t)q0 * SEQ;

  const v16bf qa0 = load_a_bf16(Qbase + 0,  DKH);
  const v16bf qa1 = load_a_bf16(Qbase + 32, DKH);

  v8f acc0 = {}, acc1 = {}, acc2 = {}, acc3 = {};
  float m_i[8], l_i[8];
  #pragma unroll
  for (int r = 0; r < 8; ++r) { m_i[r] = -3.0e38f; l_i[r] = 0.0f; }

  __bf16* ps = &pstage[wave][0];

#if USE_TDM
  if (wave == 0) {
    // prologue: V block (32x64 contiguous) + K block (64 rows x 32 keys).
    tdm_load_2d(lds_byte_addr(&vbuf[0][0]), Vbase, 32 * DKH, 1,
                SEQ * DKH, 1, SEQ * DKH);
    tdm_load_2d(lds_byte_addr(&kbuf[0][0]), Kbase, 32, DKH,
                SEQ, DKH, SEQ);
  }
  int cur = 0;
#endif

  for (int kb = 0; kb < SEQ; kb += 32) {
#if USE_TDM
    if (wave == 0) {
      if (kb + 32 < SEQ) {
        tdm_load_2d(lds_byte_addr(&vbuf[cur ^ 1][0]),
                    Vbase + (size_t)(kb + 32) * DKH, 32 * DKH, 1,
                    SEQ * DKH, 1, SEQ * DKH);
        tdm_load_2d(lds_byte_addr(&kbuf[cur ^ 1][0]),
                    Kbase + (kb + 32), 32, DKH, SEQ, DKH, SEQ);
        __builtin_amdgcn_s_wait_tensorcnt(2);   // current buffer's 2 DMAs done
      } else {
        __builtin_amdgcn_s_wait_tensorcnt(0);
      }
    }
    __syncthreads();
    const __bf16* vsrc = &vbuf[cur][0];   // [key][d], ld 64
    const __bf16* ksrc = &kbuf[cur][0];   // [d][key], ld 32
    const int vld = DKH, kld = 32, kofs = 0;
#else
    if (kb + 32 < SEQ) {
      __builtin_prefetch(Vbase + (size_t)(kb + 32) * DKH, 0, 1);
      __builtin_prefetch(Kbase + (kb + 32), 0, 1);
    }
    const __bf16* vsrc = Vbase + (size_t)kb * DKH;
    const __bf16* ksrc = Kbase;
    const int vld = DKH, kld = SEQ, kofs = kb;
#endif

    // ---- scores tile S[16 x 32] = Q(16x64) @ V^T ----
    v8f s0 = {}, s1 = {};
    s0 = wmma_bf16f32(qa0, load_bt_bf16(vsrc + 0 * vld + 0,  vld), s0);
    s0 = wmma_bf16f32(qa1, load_bt_bf16(vsrc + 0 * vld + 32, vld), s0);
    s1 = wmma_bf16f32(qa0, load_bt_bf16(vsrc + 16 * vld + 0,  vld), s1);
    s1 = wmma_bf16f32(qa1, load_bt_bf16(vsrc + 16 * vld + 32, vld), s1);

    // ---- mask + scale, row-max over the 32 keys ----
    float sc0[8], sc1[8], rmax[8];
    #pragma unroll
    for (int r = 0; r < 8; ++r) {
      const int    qr   = r + 8 * hl;
      const size_t mrow = (size_t)qr * SEQ + kb;
      sc0[r] = mbase[mrow + coln]      ? s0[r] * 0.125f : -1.0e9f;
      sc1[r] = mbase[mrow + 16 + coln] ? s1[r] * 0.125f : -1.0e9f;
      rmax[r] = fmaxf(sc0[r], sc1[r]);
    }
    #pragma unroll
    for (int xm = 1; xm <= 8; xm <<= 1)
      #pragma unroll
      for (int r = 0; r < 8; ++r)
        rmax[r] = fmaxf(rmax[r], __shfl_xor(rmax[r], xm, 32));

    // ---- online softmax ----
    float p0[8], p1[8], rsum[8], resc[8];
    #pragma unroll
    for (int r = 0; r < 8; ++r) {
      const float mn = fmaxf(m_i[r], rmax[r]);
      p0[r] = __expf(sc0[r] - mn);
      p1[r] = __expf(sc1[r] - mn);
      rsum[r] = p0[r] + p1[r];
      resc[r] = __expf(m_i[r] - mn);
      m_i[r] = mn;
    }
    #pragma unroll
    for (int xm = 1; xm <= 8; xm <<= 1)
      #pragma unroll
      for (int r = 0; r < 8; ++r)
        rsum[r] += __shfl_xor(rsum[r], xm, 32);
    #pragma unroll
    for (int r = 0; r < 8; ++r) {
      l_i[r] = l_i[r] * resc[r] + rsum[r];
      acc0[r] *= resc[r]; acc1[r] *= resc[r];
      acc2[r] *= resc[r]; acc3[r] *= resc[r];
    }

    // ---- re-swizzle P: C-layout -> A-layout via per-wave LDS tile ----
    #pragma unroll
    for (int r = 0; r < 8; ++r) {
      const int prow = r + 8 * hl;
      ps[prow * 32 + coln]      = (__bf16)p0[r];
      ps[prow * 32 + 16 + coln] = (__bf16)p1[r];
    }
    __builtin_amdgcn_wave_barrier();
    asm volatile("s_wait_dscnt 0" ::: "memory");
    const v16bf pa = load_a_bf16(ps, 32);
    __builtin_amdgcn_wave_barrier();

    // ---- ctx += P(16x32) @ K(32x64) ----
    acc0 = wmma_bf16f32(pa, load_bt_bf16(ksrc + (size_t)( 0) * kld + kofs, kld), acc0);
    acc1 = wmma_bf16f32(pa, load_bt_bf16(ksrc + (size_t)(16) * kld + kofs, kld), acc1);
    acc2 = wmma_bf16f32(pa, load_bt_bf16(ksrc + (size_t)(32) * kld + kofs, kld), acc2);
    acc3 = wmma_bf16f32(pa, load_bt_bf16(ksrc + (size_t)(48) * kld + kofs, kld), acc3);

#if USE_TDM
    __syncthreads();   // all waves done with buf `cur` before TDM reuses it
    cur ^= 1;
#endif
  }

  // ---- normalize, store context token-major [B,S,D] bf16 ----
  #pragma unroll
  for (int r = 0; r < 8; ++r) {
    const float inv  = 1.0f / l_i[r];
    const int   qrow = q0 + r + 8 * hl;
    __bf16* o = ctx + ((size_t)b * SEQ + qrow) * DMODEL + h * DKH;
    o[ 0 + coln] = (__bf16)(acc0[r] * inv);
    o[16 + coln] = (__bf16)(acc1[r] * inv);
    o[32 + coln] = (__bf16)(acc2[r] * inv);
    o[48 + coln] = (__bf16)(acc3[r] * inv);
  }
}

// ---------------------------------------------------------------------------
// Kernel 3: output projection  out = Ctx @ Wo^T + bo  (fp32 out).
// Same 32x64-per-wave tiling as the input projections.
// ---------------------------------------------------------------------------
__global__ void __launch_bounds__(256)
out_proj_kernel(const __bf16* __restrict__ Ctx, const __bf16* __restrict__ Wo,
                const float* __restrict__ bo, float* __restrict__ out) {
  const int lane = threadIdx.x & 31;
  const int wave = threadIdx.x >> 5;
  const int hl   = (lane >> 4) & 1;
  const int coln = lane & 15;
  const int m0 = blockIdx.x * 256 + wave * 32;
  const int n0 = blockIdx.y * 64;

  v8f acc[8] = {};
  for (int k = 0; k < DMODEL; k += 32) {
    const v16bf a0 = load_a_bf16(Ctx + (size_t)m0 * DMODEL + k, DMODEL);
    const v16bf a1 = load_a_bf16(Ctx + (size_t)(m0 + 16) * DMODEL + k, DMODEL);
    #pragma unroll
    for (int t = 0; t < 4; ++t) {
      const v16bf bfrag =
          load_bt_bf16(Wo + (size_t)(n0 + 16 * t) * DMODEL + k, DMODEL);
      acc[t]     = wmma_bf16f32(a0, bfrag, acc[t]);
      acc[4 + t] = wmma_bf16f32(a1, bfrag, acc[4 + t]);
    }
  }
  #pragma unroll
  for (int g = 0; g < 2; ++g) {
    #pragma unroll
    for (int t = 0; t < 4; ++t) {
      const int n = n0 + t * 16 + coln;
      const float bv = bo[n];
      #pragma unroll
      for (int r = 0; r < 8; ++r) {
        const int m = m0 + g * 16 + r + 8 * hl;
        out[(size_t)m * DMODEL + n] = acc[g * 4 + t][r] + bv;
      }
    }
  }
}

// ---------------------------------------------------------------------------
extern "C" void kernel_launch(void* const* d_in, const int* in_sizes, int n_in,
                              void* d_out, int out_size, void* d_ws, size_t ws_size,
                              hipStream_t stream) {
  const float* value = (const float*)d_in[0];
  const float* key   = (const float*)d_in[1];
  const float* query = (const float*)d_in[2];
  const int*   mask  = (const int*)  d_in[3];
  const float* Wv = (const float*)d_in[4];
  const float* bv = (const float*)d_in[5];
  const float* Wk = (const float*)d_in[6];
  const float* bk = (const float*)d_in[7];
  const float* Wq = (const float*)d_in[8];
  const float* bq = (const float*)d_in[9];
  const float* Wo = (const float*)d_in[10];
  const float* bo = (const float*)d_in[11];

  const size_t ACT = (size_t)BATCH * SEQ * DMODEL;   // 4 Mi elements
  const size_t WGT = (size_t)DMODEL * DMODEL;        // 1 Mi elements
  const size_t total = 3 * ACT + 4 * WGT + 4 * ACT;  // 32 Mi bf16 = 64 MiB
  if (ws_size < total * sizeof(__bf16)) return;

  __bf16* p = (__bf16*)d_ws;
  __bf16* Xv  = p; p += ACT;
  __bf16* Xk  = p; p += ACT;
  __bf16* Xq  = p; p += ACT;
  __bf16* Wvb = p; p += WGT;
  __bf16* Wkb = p; p += WGT;
  __bf16* Wqb = p; p += WGT;
  __bf16* Wob = p; p += WGT;
  __bf16* Vh  = p; p += ACT;   // [B,H,S,Dk]
  __bf16* Kt  = p; p += ACT;   // [B,H,Dk,S]
  __bf16* Qh  = p; p += ACT;   // [B,H,S,Dk]
  __bf16* Ctx = p; p += ACT;   // [B,S,D]

  const dim3 bCvt(256, 1, 1);
  cvt_f32_to_bf16_kernel<<<dim3((unsigned)(ACT / 8 / 256)), bCvt, 0, stream>>>(value, Xv, (int)ACT);
  cvt_f32_to_bf16_kernel<<<dim3((unsigned)(ACT / 8 / 256)), bCvt, 0, stream>>>(key,   Xk, (int)ACT);
  cvt_f32_to_bf16_kernel<<<dim3((unsigned)(ACT / 8 / 256)), bCvt, 0, stream>>>(query, Xq, (int)ACT);
  cvt_f32_to_bf16_kernel<<<dim3((unsigned)(WGT / 8 / 256)), bCvt, 0, stream>>>(Wv, Wvb, (int)WGT);
  cvt_f32_to_bf16_kernel<<<dim3((unsigned)(WGT / 8 / 256)), bCvt, 0, stream>>>(Wk, Wkb, (int)WGT);
  cvt_f32_to_bf16_kernel<<<dim3((unsigned)(WGT / 8 / 256)), bCvt, 0, stream>>>(Wq, Wqb, (int)WGT);
  cvt_f32_to_bf16_kernel<<<dim3((unsigned)(WGT / 8 / 256)), bCvt, 0, stream>>>(Wo, Wob, (int)WGT);

  const dim3 gGemm(16, 16, 1);   // M/256 x N/64
  const dim3 bGemm(256, 1, 1);
  qkv_proj_kernel<<<gGemm, bGemm, 0, stream>>>(Xv, Wvb, bv, Vh, 0);
  qkv_proj_kernel<<<gGemm, bGemm, 0, stream>>>(Xk, Wkb, bk, Kt, 1);
  qkv_proj_kernel<<<gGemm, bGemm, 0, stream>>>(Xq, Wqb, bq, Qh, 0);

  attn_kernel<<<dim3(SEQ / 64, BATCH * NHEADS, 1), dim3(128, 1, 1), 0, stream>>>(
      Qh, Kt, Vh, mask, Ctx);

  out_proj_kernel<<<gGemm, bGemm, 0, stream>>>(Ctx, Wob, bo, (float*)d_out);
}
